// GCNLayer_8787503087822
// MI455X (gfx1250) — matrix operand order repaired and checked by reading, not compiled, not run on hardware
//
#include <hip/hip_runtime.h>

#define N_NODES 100000
#define D_IN    128
#define D_OUT   256

typedef __attribute__((ext_vector_type(16))) __bf16 v16bf;
typedef __attribute__((ext_vector_type(8)))  float  v8f;

// ---------------------------------------------------------------------------
// Device-scope native FP32 atomic add (no return -> STOREcnt path).
// Guarantees global_atomic_add_f32 instead of a possible CAS-loop expansion.
// scope:SCOPE_DEV: result must be coherent across all WGPs (cross-workgroup
// accumulation into agg).
// ---------------------------------------------------------------------------
__device__ __forceinline__ void atomic_add_f32_dev(float* p, float v) {
    asm volatile("global_atomic_add_f32 %0, %1, off scope:SCOPE_DEV"
                 :: "v"(p), "v"(v)
                 : "memory");
}

// ---------------------------------------------------------------------------
// Stage 0b: Wt[n][k] = (bf16) W[k][n]   (256 x 128 bf16 = 64 KB, one shot)
// Coalesced writes; strided reads of a 128 KB L2-resident matrix.
// ---------------------------------------------------------------------------
__global__ void transpose_convert_W_kernel(const float* __restrict__ W,
                                           __bf16* __restrict__ Wt) {
    const int idx = blockIdx.x * blockDim.x + threadIdx.x;   // over D_OUT*D_IN
    if (idx >= D_IN * D_OUT) return;
    const int n = idx / D_IN;          // output row (N)
    const int k = idx % D_IN;          // output col (K), fast -> coalesced write
    Wt[idx] = (__bf16)W[(size_t)k * D_OUT + n];
}

// ---------------------------------------------------------------------------
// Stage 1: SpMM scatter  agg[row] += val * X[col]
// One wave per edge; lane l moves float4 [4l..4l+3] (512 B/row, coalesced).
// X (51.2 MB) is L2-resident (192 MB L2) so random gathers hit L2.
// ---------------------------------------------------------------------------
__global__ void spmm_scatter_kernel(const float* __restrict__ X,
                                    const int*   __restrict__ rows,
                                    const int*   __restrict__ cols,
                                    const float* __restrict__ vals,
                                    float*       __restrict__ agg,
                                    int nEdges) {
    const int gid  = blockIdx.x * blockDim.x + threadIdx.x;
    const int e    = gid >> 5;
    const int lane = gid & 31;
    if (e >= nEdges) return;                 // wave-uniform guard

    const int   r = rows[e];                 // wave-uniform
    const int   c = cols[e];
    const float v = vals[e];

    const float4 x = *((const float4*)(X + (size_t)c * D_IN) + lane);
    float* dst = agg + (size_t)r * D_IN + lane * 4;
    atomic_add_f32_dev(dst + 0, v * x.x);
    atomic_add_f32_dev(dst + 1, v * x.y);
    atomic_add_f32_dev(dst + 2, v * x.z);
    atomic_add_f32_dev(dst + 3, v * x.w);
}

// ---------------------------------------------------------------------------
// Stage 2: out = relu(agg @ W + b) + last_hidden via v_wmma_f32_16x16x32_bf16.
// Wave computes a 32(M) x 16(N) tile: 2 accumulators share one B fragment per
// K-step (4 K-steps of 32). M = 6250*16, N = 16*16, K = 4*32 exact (no masks;
// EXEC all-1s for WMMA).
//
// Fragments (contiguous k order, same for A and B), lane l, g = l>>4:
//   A: a[j]   = A[l&15][k0 + g*16 + j]     j=0..7
//      a[8+j] = A[l&15][k0 + g*16 + 8 + j] j=0..7
//   B: b[j]   = W[k0 + g*16 + j][l&15 of tile]  -> Wt[col][k0 + g*16 + j]
//      (USE_WT: 16 contiguous bf16 = 2x global_load_b128 per K-step)
//   C/D: VGPR r -> (M = r + g*8, N = l&15)
// ---------------------------------------------------------------------------
template <bool USE_WT>
__global__ void gemm_bias_relu_res_kernel(const float*  __restrict__ agg,
                                          const float*  __restrict__ W,
                                          const __bf16* __restrict__ Wt,
                                          const float*  __restrict__ bias,
                                          const float*  __restrict__ last_hidden,
                                          float*        __restrict__ out) {
    const int wave   = (blockIdx.x * blockDim.x + threadIdx.x) >> 5;
    const int lane   = threadIdx.x & 31;
    const int tileN  = wave & 15;          // 16 N-tiles
    const int tileM2 = wave >> 4;          // 3125 double-M-tiles
    if (tileM2 >= (N_NODES / 32)) return;  // wave-uniform; EXEC stays full

    const int row = lane & 15;
    const int grp = lane >> 4;             // 0 or 1
    const int col = tileN * 16 + row;      // N index for B/C/D fragments

    const int    mbase = tileM2 * 32;
    const float* A0    = agg + (size_t)(mbase + row) * D_IN;        // M-tile 0
    const float* A1    = A0 + (size_t)16 * D_IN;                    // M-tile 1
    const __bf16* Bt   = Wt + (size_t)col * D_IN + grp * 16;        // Wt[col][.]

    v8f c0 = {};
    v8f c1 = {};

    #pragma unroll
    for (int k0 = 0; k0 < D_IN; k0 += 32) {
        v16bf a0, a1, b;

        const float* ap0 = A0 + k0 + grp * 16;
        const float* ap1 = A0 + k0 + 8 + grp * 16;
        const float* aq0 = A1 + k0 + grp * 16;
        const float* aq1 = A1 + k0 + 8 + grp * 16;
        #pragma unroll
        for (int j = 0; j < 8; ++j) {
            a0[j]     = (__bf16)ap0[j];
            a0[8 + j] = (__bf16)ap1[j];
            a1[j]     = (__bf16)aq0[j];
            a1[8 + j] = (__bf16)aq1[j];
        }

        if (USE_WT) {
            // 16 contiguous bf16 per lane -> 2x b128
            #pragma unroll
            for (int j = 0; j < 16; ++j)
                b[j] = Bt[k0 + j];
        } else {
            const float* bp = W + (size_t)(k0 + grp * 16) * D_OUT + col;
            #pragma unroll
            for (int j = 0; j < 16; ++j)
                b[j] = (__bf16)bp[(size_t)j * D_OUT];
        }

        // (neg_a, A, neg_b, B, c_mod, C, reuse_a, reuse_b)
        c0 = __builtin_amdgcn_wmma_f32_16x16x32_bf16(false, a0, false, b,
                                                     (short)0, c0, false, false);
        c1 = __builtin_amdgcn_wmma_f32_16x16x32_bf16(false, a1, false, b,
                                                     (short)0, c1, false, false);
    }

    // Fused epilogue: + bias, relu, + residual
    const float bv = bias[col];
    #pragma unroll
    for (int r = 0; r < 8; ++r) {
        const int m0 = mbase + grp * 8 + r;         // M-tile 0 row
        const int m1 = m0 + 16;                     // M-tile 1 row
        float v0 = c0[r] + bv;
        float v1 = c1[r] + bv;
        v0 = v0 > 0.0f ? v0 : 0.0f;
        v1 = v1 > 0.0f ? v1 : 0.0f;
        out[(size_t)m0 * D_OUT + col] = v0 + last_hidden[(size_t)m0 * D_OUT + col];
        out[(size_t)m1 * D_OUT + col] = v1 + last_hidden[(size_t)m1 * D_OUT + col];
    }
}

// ---------------------------------------------------------------------------
// Launch: memset(agg) -> transpose W -> SpMM scatter -> WMMA GEMM (on stream)
// Workspace layout: [ agg: 100000*128 f32 = 51.2 MB | Wt: 256*128 bf16 = 64 KB ]
// ---------------------------------------------------------------------------
extern "C" void kernel_launch(void* const* d_in, const int* in_sizes, int n_in,
                              void* d_out, int out_size, void* d_ws, size_t ws_size,
                              hipStream_t stream) {
    const float* X    = (const float*)d_in[0];   // [N_NODES, D_IN]
    const int*   er   = (const int*)  d_in[1];   // [E]
    const int*   ec   = (const int*)  d_in[2];   // [E]
    const float* ev   = (const float*)d_in[3];   // [E]
    const float* W    = (const float*)d_in[4];   // [D_IN, D_OUT]
    const float* bias = (const float*)d_in[5];   // [D_OUT]
    const float* lh   = (const float*)d_in[6];   // [N_NODES, D_OUT]
    float*       out  = (float*)d_out;           // [N_NODES, D_OUT]

    const size_t aggBytes = (size_t)N_NODES * D_IN * sizeof(float);
    const size_t wtBytes  = (size_t)D_IN * D_OUT * sizeof(__bf16);
    float*  agg = (float*)d_ws;
    __bf16* Wt  = (__bf16*)((char*)d_ws + aggBytes);
    const bool useWt = (ws_size >= aggBytes + wtBytes);

    const int E = in_sizes[1];

    // Zero the aggregation buffer (graph-capture safe).
    hipMemsetAsync(agg, 0, aggBytes, stream);

    // One-shot W transpose+convert to bf16 (64 KB), if workspace allows.
    if (useWt) {
        const int n = D_IN * D_OUT;
        transpose_convert_W_kernel<<<(n + 255) / 256, 256, 0, stream>>>(W, Wt);
    }

    // SpMM: one wave per edge.
    {
        const long long threads = (long long)E * 32;
        const int block = 256;
        const int grid  = (int)((threads + block - 1) / block);
        spmm_scatter_kernel<<<grid, block, 0, stream>>>(X, er, ec, ev, agg, E);
    }

    // GEMM: 50000 waves (3125 x 16 tiles of 32x16), 8 waves/block.
    {
        const int block = 256;
        const int grid  = (N_NODES / 32) * 16 / (block / 32);   // 6250
        if (useWt)
            gemm_bias_relu_res_kernel<true><<<grid, block, 0, stream>>>(
                agg, W, Wt, bias, lh, out);
        else
            gemm_bias_relu_res_kernel<false><<<grid, block, 0, stream>>>(
                agg, W, Wt, bias, lh, out);
    }
}
